// SecondStage_33294586479281
// MI455X (gfx1250) — matrix operand order repaired and checked by reading, not compiled, not run on hardware
//
#include <hip/hip_runtime.h>
#include <math.h>

// ---- problem dimensions (fixed by the reference) ----
#define B_DIM 1024
#define I_DIM 256
#define H_DIM 256
#define T_DIM 64
#define TM1   63

typedef __attribute__((ext_vector_type(16))) __bf16 v16bf;
typedef __attribute__((ext_vector_type(8)))  __bf16 v8bf;
typedef __attribute__((ext_vector_type(8)))  float  v8f;

__device__ __forceinline__ float fast_tanh(float x) {
#if __has_builtin(__builtin_amdgcn_tanhf)
    return __builtin_amdgcn_tanhf(x);
#else
    return tanhf(x);
#endif
}
__device__ __forceinline__ float sigmoidf(float x) {
    return 1.0f / (1.0f + __expf(-x));
}

// Zero LSTM state and the bf16 h-part of the fused GEMM input.
__global__ void init_state_kernel(float* __restrict__ h, float* __restrict__ c,
                                  __bf16* __restrict__ xbf) {
    int idx = blockIdx.x * blockDim.x + threadIdx.x;   // 0 .. B*H-1
    h[idx] = 0.0f;
    c[idx] = 0.0f;
    int b = idx >> 8, j = idx & 255;
    xbf[b * 512 + 256 + j] = (__bf16)0.0f;
}

// Build bf16 fused weight matrix Wc (4H x 2H): Wc[j][k] = k<256 ? W_ih[j][k] : W_hh[j][k-256]
__global__ void convert_w_kernel(const float* __restrict__ W_ih,
                                 const float* __restrict__ W_hh,
                                 __bf16* __restrict__ wcbf) {
    int idx = blockIdx.x * blockDim.x + threadIdx.x;   // 0 .. 1024*512-1
    int j = idx >> 9, k = idx & 511;
    float v = (k < 256) ? W_ih[j * 256 + k] : W_hh[j * 256 + (k - 256)];
    wcbf[idx] = (__bf16)v;
}

// pre_fs[b,i,u] = b_fs[u] + sum_t FS[b,t,i] * W_fs[t,u]   (t = 0..62, u = 0..63)
// One block per batch, 256 threads = i dimension; FS column cached in registers.
__global__ void prefs_kernel(const float* __restrict__ FS,
                             const float* __restrict__ W_fs,
                             const float* __restrict__ b_fs,
                             float* __restrict__ pre_fs) {
    __shared__ float wfs_s[64 * 64];
    __shared__ float bfs_s[64];
    int b = blockIdx.x, tid = threadIdx.x;
    for (int n = tid; n < 64 * 64; n += 256) wfs_s[n] = W_fs[n];
    if (tid < 64) bfs_s[tid] = b_fs[tid];
    __syncthreads();

    float f[TM1];
    const float* fsb = FS + (size_t)b * TM1 * I_DIM + tid;
#pragma unroll
    for (int t = 0; t < TM1; ++t) f[t] = fsb[t * I_DIM];

    float* outp = pre_fs + ((size_t)b * I_DIM + tid) * T_DIM;
    for (int u = 0; u < T_DIM; ++u) {
        float acc = bfs_s[u];
#pragma unroll
        for (int t = 0; t < TM1; ++t) acc = fmaf(f[t], wfs_s[t * 64 + u], acc);
        outp[u] = acc;
    }
}

// Per-step attention: score_h, tanh-attention reduction, softmax over I, w_in.
// One block per batch element, 256 threads = I dimension.
__global__ void attn_kernel(const float* __restrict__ pre_fs,
                            const float* __restrict__ FS,
                            const float* __restrict__ y_hist,
                            const float* __restrict__ W_h,
                            const float* __restrict__ b_h,
                            const float* __restrict__ W_fs,
                            const float* __restrict__ W_attn,
                            const float* __restrict__ b_attn,
                            const float* __restrict__ h,
                            const float* __restrict__ c,
                            float* __restrict__ iw_out,
                            __bf16* __restrict__ xbf,
                            int t) {
    __shared__ float sh[64];
    __shared__ float wat[64];
    __shared__ float part[4 * 64];
    __shared__ float red[256];
    int b = blockIdx.x, tid = threadIdx.x;
    int u = tid & 63, pc = tid >> 6;

    // score_h[u] = sum_k [h|c][k] * W_h[k][u]  (4-way K split across the block)
    float acc = 0.0f;
    const float* hb = h + b * H_DIM;
    const float* cb = c + b * H_DIM;
    for (int k = pc * 128; k < (pc + 1) * 128; ++k) {
        float hcv = (k < H_DIM) ? hb[k] : cb[k - H_DIM];
        acc = fmaf(hcv, W_h[k * 64 + u], acc);
    }
    part[pc * 64 + u] = acc;
    if (tid < 64) wat[tid] = W_attn[tid];
    __syncthreads();
    if (tid < 64) {
        float y = y_hist[b * TM1 + t];
        sh[tid] = part[tid] + part[64 + tid] + part[128 + tid] + part[192 + tid]
                + b_h[tid] + y * W_fs[63 * 64 + tid];
    }
    __syncthreads();

    // attn_i = b_attn + sum_u tanh(sh[u] + pre_fs[b,i,u]) * W_attn[u]
    const float* pf = pre_fs + ((size_t)b * I_DIM + tid) * T_DIM;
    float at = b_attn[0];
#pragma unroll
    for (int uu = 0; uu < 64; ++uu)
        at = fmaf(fast_tanh(sh[uu] + pf[uu]), wat[uu], at);

    // softmax over the 256 i-values of this batch element
    red[tid] = at;
    __syncthreads();
    for (int s = 128; s > 0; s >>= 1) {
        if (tid < s) red[tid] = fmaxf(red[tid], red[tid + s]);
        __syncthreads();
    }
    float m = red[0];
    __syncthreads();
    float e = __expf(at - m);
    red[tid] = e;
    __syncthreads();
    for (int s = 128; s > 0; s >>= 1) {
        if (tid < s) red[tid] += red[tid + s];
        __syncthreads();
    }
    float a = e / red[0];

    float fsv = FS[((size_t)b * TM1 + t) * I_DIM + tid];
    float w = a * fsv;
    iw_out[((size_t)b * TM1 + t) * I_DIM + tid] = w;
    xbf[b * 512 + tid] = (__bf16)w;
}

// gates = [w_in | h](B x 512, bf16) @ Wc^T(512 x 1024, bf16) + (b_ih + b_hh), fp32 accumulate.
// 256 threads = 8 waves; each wave computes a 16(M) x 64(N) tile with a K-loop of 16
// v_wmma_f32_16x16x32_bf16 steps (4 per iter, A-fragment reused across the 4 N-subtiles).
__global__ void gates_wmma_kernel(const __bf16* __restrict__ xbf,
                                  const __bf16* __restrict__ wcbf,
                                  const float* __restrict__ b_ih,
                                  const float* __restrict__ b_hh,
                                  float* __restrict__ gates) {
    int wave = threadIdx.x >> 5;
    int lane = threadIdx.x & 31;
    int kg = lane >> 4;        // half-wave selects the K-subgroup
    int lh = lane & 15;

    int m0 = (blockIdx.x * 8 + wave) * 16;   // batch-row tile
    int n0 = blockIdx.y * 64;                // gate-column tile

    v8f acc[4] = {};

    const __bf16* xr0 = xbf + (size_t)(m0 + lh) * 512 + kg * 16;
    for (int kk = 0; kk < 512; kk += 32) {
        // A fragment: lane (m=lh, kgroup=kg) holds K = kk+kg*16 .. +15
        v8bf lo = *reinterpret_cast<const v8bf*>(xr0 + kk);
        v8bf hi = *reinterpret_cast<const v8bf*>(xr0 + kk + 8);
        v16bf afrag;
#pragma unroll
        for (int j = 0; j < 8; ++j) { afrag[j] = lo[j]; afrag[8 + j] = hi[j]; }

#pragma unroll
        for (int nt = 0; nt < 4; ++nt) {
            // B fragment: B[k][n] = Wc[n][k]; lane n=n0+nt*16+lh holds K = kk+16*kg .. +15
            int n = n0 + nt * 16 + lh;
            v16bf bfrag = *reinterpret_cast<const v16bf*>(wcbf + (size_t)n * 512 + kk + kg * 16);
            acc[nt] = __builtin_amdgcn_wmma_f32_16x16x32_bf16(
                false, afrag, false, bfrag, (short)0, acc[nt], false, false);
        }
    }

#pragma unroll
    for (int nt = 0; nt < 4; ++nt) {
        int col = n0 + nt * 16 + lh;
        float bias = b_ih[col] + b_hh[col];
#pragma unroll
        for (int r = 0; r < 8; ++r) {
            int row = m0 + r + 8 * kg;       // D layout: vgpr r -> M = r (+8 for upper lanes)
            gates[(size_t)row * 1024 + col] = acc[nt][r] + bias;
        }
    }
}

// LSTM elementwise: c,h update; write h into input_encoded and bf16 copy into xbf.
__global__ void lstm_kernel(const float* __restrict__ gates,
                            float* __restrict__ h, float* __restrict__ c,
                            float* __restrict__ ie_out, __bf16* __restrict__ xbf,
                            int t) {
    int idx = blockIdx.x * blockDim.x + threadIdx.x;   // 0 .. B*H-1
    int b = idx >> 8, j = idx & 255;
    const float* g = gates + (size_t)b * 1024;
    float ig = sigmoidf(g[j]);
    float fg = sigmoidf(g[256 + j]);
    float gg = fast_tanh(g[512 + j]);
    float og = sigmoidf(g[768 + j]);
    float cn = fg * c[idx] + ig * gg;
    float hn = og * fast_tanh(cn);
    c[idx] = cn;
    h[idx] = hn;
    ie_out[((size_t)b * TM1 + t) * H_DIM + j] = hn;
    xbf[b * 512 + 256 + j] = (__bf16)hn;
}

extern "C" void kernel_launch(void* const* d_in, const int* in_sizes, int n_in,
                              void* d_out, int out_size, void* d_ws, size_t ws_size,
                              hipStream_t stream) {
    (void)in_sizes; (void)n_in; (void)out_size; (void)ws_size;
    const float* FS     = (const float*)d_in[0];   // (B, 63, I)
    const float* y_h    = (const float*)d_in[1];   // (B, 63)
    const float* W_h    = (const float*)d_in[2];   // (512, 64)
    const float* b_h    = (const float*)d_in[3];   // (64,)
    const float* W_fs   = (const float*)d_in[4];   // (64, 64)
    const float* b_fs   = (const float*)d_in[5];   // (64,)
    const float* W_attn = (const float*)d_in[6];   // (64, 1)
    const float* b_attn = (const float*)d_in[7];   // (1,)
    const float* W_ih   = (const float*)d_in[8];   // (1024, 256)
    const float* W_hh   = (const float*)d_in[9];   // (1024, 256)
    const float* b_ih   = (const float*)d_in[10];  // (1024,)
    const float* b_hh   = (const float*)d_in[11];  // (1024,)

    float* iw = (float*)d_out;                                  // (B, 63, I)
    float* ie = iw + (size_t)B_DIM * TM1 * I_DIM;               // (B, 63, H)

    char* ws = (char*)d_ws;
    size_t off = 0;
    auto alloc = [&](size_t bytes) {
        void* p = ws + off;
        off = (off + bytes + 255) & ~(size_t)255;
        return p;
    };
    float*  pre_fs = (float*) alloc((size_t)B_DIM * I_DIM * T_DIM * sizeof(float)); // 64 MB
    float*  gates  = (float*) alloc((size_t)B_DIM * 1024 * sizeof(float));          // 4 MB
    float*  hbuf   = (float*) alloc((size_t)B_DIM * H_DIM * sizeof(float));
    float*  cbuf   = (float*) alloc((size_t)B_DIM * H_DIM * sizeof(float));
    __bf16* xbf    = (__bf16*)alloc((size_t)B_DIM * 512 * sizeof(__bf16));
    __bf16* wcbf   = (__bf16*)alloc((size_t)1024 * 512 * sizeof(__bf16));

    init_state_kernel<<<B_DIM * H_DIM / 256, 256, 0, stream>>>(hbuf, cbuf, xbf);
    convert_w_kernel<<<1024 * 512 / 256, 256, 0, stream>>>(W_ih, W_hh, wcbf);
    prefs_kernel<<<B_DIM, 256, 0, stream>>>(FS, W_fs, b_fs, pre_fs);

    for (int t = 0; t < TM1; ++t) {
        attn_kernel<<<B_DIM, 256, 0, stream>>>(pre_fs, FS, y_h, W_h, b_h, W_fs,
                                               W_attn, b_attn, hbuf, cbuf, iw, xbf, t);
        gates_wmma_kernel<<<dim3(8, 16), 256, 0, stream>>>(xbf, wcbf, b_ih, b_hh, gates);
        lstm_kernel<<<B_DIM * H_DIM / 256, 256, 0, stream>>>(gates, hbuf, cbuf, ie, xbf, t);
    }
}
